// MultiheadSelfAttention_74964359185071
// MI455X (gfx1250) — matrix operand order repaired and checked by reading, not compiled, not run on hardware
//
#include <hip/hip_runtime.h>
#include <cstdint>

// ---------------------------------------------------------------------------
// MI455X (gfx1250) multi-head self-attention, bf16 WMMA + fp32 accumulate.
//   k1: qkv = x @ qkv_w^T (+RoPE on q,k) -> Qb[b,h,s,d], Kb[b,h,s,d], Vt[b,h,d,s] (bf16)
//   k2: flash attention (64 q-rows x 64-key tiles) -> An[b,s,h*64+d] (bf16)
//   k3: out = An @ o_w^T (fp32)
// ---------------------------------------------------------------------------

#define S_LEN 2048
#define DM    1024
#define NHD   16
#define DK    64

typedef __bf16 v16bf __attribute__((ext_vector_type(16)));
typedef __bf16 v8bf  __attribute__((ext_vector_type(8)));
typedef __bf16 v4bf  __attribute__((ext_vector_type(4)));
typedef float  v8f   __attribute__((ext_vector_type(8)));

// ---- async global->LDS staging (CDNA5 GLOBAL_LOAD_ASYNC_TO_LDS_B128) ------
#if defined(__HIP_DEVICE_COMPILE__) &&                                         \
    __has_builtin(__builtin_amdgcn_global_load_async_to_lds_b128) &&           \
    __has_builtin(__builtin_amdgcn_s_wait_asynccnt)
#define ASYNC_LDS 1
// Builtin parameter type (from clang diagnostic): pointer to
// __attribute__((vector_size(16))) int, global (AS1) source / LDS (AS3) dest.
typedef int v4i_raw __attribute__((vector_size(4 * sizeof(int))));
typedef __attribute__((address_space(1))) v4i_raw* as1_v4i;
typedef __attribute__((address_space(3))) v4i_raw* as3_v4i;
__device__ __forceinline__ void async_copy16(const void* g, void* l) {
  // generic LDS pointer: low 32 bits are the LDS byte address (ISA 10.2)
  __builtin_amdgcn_global_load_async_to_lds_b128(
      (as1_v4i)(uintptr_t)g, (as3_v4i)(uint32_t)(uintptr_t)l, 0, 0);
}
__device__ __forceinline__ void async_wait0() {
  __builtin_amdgcn_s_wait_asynccnt(0);
}
#else
#define ASYNC_LDS 0
__device__ __forceinline__ void async_copy16(const void* g, void* l) {
  *(uint4*)l = *(const uint4*)g;
}
__device__ __forceinline__ void async_wait0() {}
#endif

// D = A(16x32 bf16) * B(32x16 bf16) + C(16x16 f32)
__device__ __forceinline__ v8f wmma_bf16(v16bf a, v16bf b, v8f c) {
  return __builtin_amdgcn_wmma_f32_16x16x32_bf16(
      /*neg_a=*/false, a, /*neg_b=*/false, b,
      /*c_mod=*/(short)0, c, /*reuse_a=*/false, /*reuse_b=*/false);
}

// A-fragment: lane L holds row; element i -> K = 16*(i/8) + 8*(L>=16) + i%8.
__device__ __forceinline__ v16bf frag_a(const __bf16* t, int stride, int row,
                                        int col, int hi) {
  const __bf16* p = t + row * stride + col + (hi << 3);
  v8bf lo = *(const v8bf*)p;
  v8bf hh = *(const v8bf*)(p + 16);
  return __builtin_shufflevector(lo, hh, 0, 1, 2, 3, 4, 5, 6, 7,
                                 8, 9, 10, 11, 12, 13, 14, 15);
}

// B-fragment: lane L holds column N=L%16; element i -> K = 16*(i/8) + 8*(L>=16) + i%8.
// `row` = column index N, reading a row of the K-major (transposed) tile.
__device__ __forceinline__ v16bf frag_b(const __bf16* t, int stride, int row,
                                        int col, int hi) {
  const __bf16* p = t + row * stride + col + (hi << 3);
  v8bf lo = *(const v8bf*)p;
  v8bf hh = *(const v8bf*)(p + 16);
  return __builtin_shufflevector(lo, hh, 0, 1, 2, 3, 4, 5, 6, 7,
                                 8, 9, 10, 11, 12, 13, 14, 15);
}

// ---------------------------------------------------------------------------
// Kernel 1: qkv projection + RoPE. Grid (48, 32), block 256 (8 waves).
// Block tile 128(M) x 64(N); wave w: rows (w&3)*32, cols (w>>2)*32 -> 4 WMMAs.
// ---------------------------------------------------------------------------
__global__ void __launch_bounds__(256)
qkv_rope_kernel(const float* __restrict__ x, const int* __restrict__ tok,
                const float* __restrict__ qkv_w,
                unsigned short* __restrict__ Qb_, unsigned short* __restrict__ Kb_,
                unsigned short* __restrict__ Vt_) {
  __bf16* Qb = (__bf16*)Qb_;
  __bf16* Kb = (__bf16*)Kb_;
  __bf16* Vt = (__bf16*)Vt_;

  __shared__ alignas(16) __bf16 Xt[128 * 32];
  __shared__ alignas(16) __bf16 Wt[64 * 32];

  const int tid = threadIdx.x;
  const int wave = tid >> 5, ln = tid & 31, l16 = ln & 15, hi = ln >> 4;
  const int m0 = blockIdx.y * 128;  // flattened b*S+s
  const int n0 = blockIdx.x * 64;   // qkv feature
  const int mi = wave & 3, ns = wave >> 2;

  v8f acc[4] = {};  // [ai*2+bi]

  for (int kk = 0; kk < DM; kk += 32) {
    __syncthreads();
#pragma unroll
    for (int i = 0; i < 4; ++i) {   // X tile: 128x32 f32 -> bf16 (1024 float4)
      const int slot = tid + i * 256;
      const int row = slot >> 3, c4 = (slot & 7) << 2;
      const float* src = x + (size_t)(m0 + row) * DM + kk + c4;
      float4 vx = *(const float4*)src;
      __builtin_prefetch(src + 32, 0, 1);   // next K-tile -> global_prefetch
      v4bf bx;
      bx[0] = (__bf16)vx.x; bx[1] = (__bf16)vx.y;
      bx[2] = (__bf16)vx.z; bx[3] = (__bf16)vx.w;
      *(v4bf*)(Xt + row * 32 + c4) = bx;
    }
#pragma unroll
    for (int i = 0; i < 2; ++i) {   // W tile: 64x32 f32 -> bf16 (512 float4)
      const int slot = tid + i * 256;
      const int row = slot >> 3, c4 = (slot & 7) << 2;
      float4 vw = *(const float4*)(qkv_w + (size_t)(n0 + row) * DM + kk + c4);
      v4bf bw;
      bw[0] = (__bf16)vw.x; bw[1] = (__bf16)vw.y;
      bw[2] = (__bf16)vw.z; bw[3] = (__bf16)vw.w;
      *(v4bf*)(Wt + row * 32 + c4) = bw;
    }
    __syncthreads();

    const v16bf a0 = frag_a(Xt, 32, mi * 32 + l16, 0, hi);
    const v16bf a1 = frag_a(Xt, 32, mi * 32 + 16 + l16, 0, hi);
    const v16bf b0 = frag_b(Wt, 32, ns * 32 + l16, 0, hi);
    const v16bf b1 = frag_b(Wt, 32, ns * 32 + 16 + l16, 0, hi);
    acc[0] = wmma_bf16(a0, b0, acc[0]);
    acc[1] = wmma_bf16(a0, b1, acc[1]);
    acc[2] = wmma_bf16(a1, b0, acc[2]);
    acc[3] = wmma_bf16(a1, b1, acc[3]);
  }

  // Epilogue: C layout -> lane holds col N=l16, rows r + 8*hi.
#pragma unroll
  for (int bi = 0; bi < 2; ++bi) {
    const int o = n0 + ns * 32 + bi * 16 + l16;  // region uniform across wave
    if (o < 2 * DM) {                            // q or k: apply RoPE
      const int oq = o & (DM - 1);
      const int h = oq >> 6, d = oq & 63;
      __bf16* dst = (o < DM) ? Qb : Kb;
      const float freq = __powf(10000.0f, -(float)(2 * (d >> 1)) * (1.0f / 64.0f));
      const int odd = d & 1;
#pragma unroll
      for (int ai = 0; ai < 2; ++ai)
#pragma unroll
        for (int r = 0; r < 8; ++r) {
          const int M = m0 + mi * 32 + ai * 16 + r + (hi << 3);
          const int b = M >> 11, s = M & (S_LEN - 1);
          const float v = acc[ai * 2 + bi][r];
          const float partner = __shfl_xor(v, 1);  // even<->odd feature column
          const float pos = (float)tok[s];
          float sn, cs;
          __sincosf(pos * freq, &sn, &cs);
          const float res = odd ? (partner * sn + v * cs)
                                : (v * cs - partner * sn);
          dst[((size_t)(b * NHD + h) * S_LEN + s) * DK + d] = (__bf16)res;
        }
    } else {                                     // v: store transposed [bh][d][s]
      const int ov = o - 2 * DM;
      const int h = ov >> 6, d = ov & 63;
#pragma unroll
      for (int ai = 0; ai < 2; ++ai)
#pragma unroll
        for (int r = 0; r < 8; ++r) {
          const int M = m0 + mi * 32 + ai * 16 + r + (hi << 3);
          const int b = M >> 11, s = M & (S_LEN - 1);
          Vt[((size_t)(b * NHD + h) * DK + d) * S_LEN + s] =
              (__bf16)acc[ai * 2 + bi][r];
        }
    }
  }
}

// ---------------------------------------------------------------------------
// Kernel 2: flash attention. Grid (S/64, B*H), block 128 (4 waves).
// Each wave owns 16 q-rows; streams 64-key tiles (8 QK^T + 8 PV WMMAs/iter)
// with online softmax. K/V/Q tiles staged via async global->LDS.
// ---------------------------------------------------------------------------
__global__ void __launch_bounds__(128)
attn_kernel(const unsigned short* __restrict__ Qb_,
            const unsigned short* __restrict__ Kb_,
            const unsigned short* __restrict__ Vt_,
            unsigned short* __restrict__ An_) {
  const __bf16* Qb = (const __bf16*)Qb_;
  const __bf16* Kb = (const __bf16*)Kb_;
  const __bf16* Vt = (const __bf16*)Vt_;
  __bf16* An = (__bf16*)An_;

  __shared__ alignas(16) __bf16 Qt[64 * 64];      // 64 q-rows x 64 d
  __shared__ alignas(16) __bf16 Kt[64 * 64];      // 64 k-rows x 64 d
  __shared__ alignas(16) __bf16 Vtt[64 * 64];     // 64 d x 64 k (transposed V)
  __shared__ alignas(16) __bf16 Pst[4][16 * 64];  // per-wave P staging

  const int tid = threadIdx.x;
  const int wave = tid >> 5, ln = tid & 31, l16 = ln & 15, hi = ln >> 4;
  const int qb = blockIdx.x * 64;
  const int bh = blockIdx.y;
  const int b = bh >> 4, h = bh & 15;
  const size_t headoff = (size_t)bh * S_LEN * DK;

  { // Q tile contiguous in [bh][s][d]: async 16B copies (512 total)
    const __bf16* src = Qb + headoff + (size_t)qb * DK;
    for (int i = tid; i < 512; i += 128)
      async_copy16(src + i * 8, Qt + i * 8);
  }

  v8f acc[4] = {};
  float m[8], l[8];
#pragma unroll
  for (int r = 0; r < 8; ++r) { m[r] = -1e30f; l[r] = 0.0f; }

  const int qrow_lo = qb + wave * 16;
  const int qmax = qrow_lo + 15;

  for (int kb = 0; kb < qb + 64; kb += 64) {
    __syncthreads();
    { // K tile (contiguous 64x64) + V^T tile (128B row segments, stride S)
      const __bf16* ks = Kb + headoff + (size_t)kb * DK;
      for (int i = tid; i < 512; i += 128)
        async_copy16(ks + i * 8, Kt + i * 8);
      const __bf16* vs = Vt + (size_t)bh * DK * S_LEN + kb;
      for (int i = tid; i < 512; i += 128) {
        const int row = i >> 3, c = (i & 7) << 3;
        async_copy16(vs + (size_t)row * S_LEN + c, Vtt + row * 64 + c);
      }
    }
    async_wait0();       // ASYNCcnt not covered by the workgroup fence
    __syncthreads();

    if (kb <= qmax) {    // wave-uniform: skip fully-masked tiles
      v16bf aq[2];
#pragma unroll
      for (int kc = 0; kc < 2; ++kc)
        aq[kc] = frag_a(Qt, 64, wave * 16 + l16, kc * 32, hi);

      v8f sc[4] = {};
#pragma unroll
      for (int nh = 0; nh < 4; ++nh)
#pragma unroll
        for (int kc = 0; kc < 2; ++kc)
          sc[nh] = wmma_bf16(aq[kc],
                             frag_b(Kt, 64, nh * 16 + l16, kc * 32, hi),
                             sc[nh]);

      // scale + causal mask + row max
      float rmax[8];
#pragma unroll
      for (int r = 0; r < 8; ++r) {
        const int qrow = qrow_lo + r + (hi << 3);
        float mx = -1e30f;
#pragma unroll
        for (int nh = 0; nh < 4; ++nh) {
          const int kpos = kb + nh * 16 + l16;
          float v = sc[nh][r] * 0.125f;            // 1/sqrt(64)
          if (kpos > qrow) v = -1e30f;
          sc[nh][r] = v;
          mx = fmaxf(mx, v);
        }
        rmax[r] = mx;
      }
#pragma unroll
      for (int r = 0; r < 8; ++r)
#pragma unroll
        for (int off = 1; off < 16; off <<= 1)
          rmax[r] = fmaxf(rmax[r], __shfl_xor(rmax[r], off));

      float sf[8], rsum[8];
#pragma unroll
      for (int r = 0; r < 8; ++r) {
        const float mnew = fmaxf(m[r], rmax[r]);
        sf[r] = __expf(m[r] - mnew);
        m[r] = mnew;
        float rs = 0.0f;
#pragma unroll
        for (int nh = 0; nh < 4; ++nh) {
          const float p = __expf(sc[nh][r] - mnew);
          sc[nh][r] = p;
          rs += p;
        }
        rsum[r] = rs;
      }
#pragma unroll
      for (int r = 0; r < 8; ++r)
#pragma unroll
        for (int off = 1; off < 16; off <<= 1)
          rsum[r] += __shfl_xor(rsum[r], off);
#pragma unroll
      for (int r = 0; r < 8; ++r) l[r] = l[r] * sf[r] + rsum[r];
#pragma unroll
      for (int j = 0; j < 4; ++j)
#pragma unroll
        for (int r = 0; r < 8; ++r) acc[j][r] *= sf[r];

      // P: C layout -> LDS -> A layout (wave-private, in-order LDS)
      __bf16* ps = Pst[wave];
#pragma unroll
      for (int r = 0; r < 8; ++r) {
        const int M = r + (hi << 3);
#pragma unroll
        for (int nh = 0; nh < 4; ++nh)
          ps[M * 64 + nh * 16 + l16] = (__bf16)sc[nh][r];
      }
      v16bf ap[2];
#pragma unroll
      for (int kc = 0; kc < 2; ++kc)
        ap[kc] = frag_a(ps, 64, l16, kc * 32, hi);
#pragma unroll
      for (int j = 0; j < 4; ++j)
#pragma unroll
        for (int kc = 0; kc < 2; ++kc)
          acc[j] = wmma_bf16(ap[kc],
                             frag_b(Vtt, 64, j * 16 + l16, kc * 32, hi),
                             acc[j]);
    }
  }

  // normalize + store attn tile as [b, s, h*64+d] bf16 (row-major over D)
#pragma unroll
  for (int j = 0; j < 4; ++j)
#pragma unroll
    for (int r = 0; r < 8; ++r) {
      const int s = qrow_lo + r + (hi << 3);
      const int col = h * DK + j * 16 + l16;
      An[(size_t)(b * S_LEN + s) * DM + col] = (__bf16)(acc[j][r] / l[r]);
    }
}

// ---------------------------------------------------------------------------
// Kernel 3: output projection. Grid (16, 32), block 256 (8 waves).
// Block tile 128(M) x 64(N); 4 WMMAs per wave per K-step.
// ---------------------------------------------------------------------------
__global__ void __launch_bounds__(256)
oproj_kernel(const unsigned short* __restrict__ An_,
             const float* __restrict__ ow, float* __restrict__ out) {
  const __bf16* An = (const __bf16*)An_;
  __shared__ alignas(16) __bf16 At[128 * 32];
  __shared__ alignas(16) __bf16 Wt[64 * 32];

  const int tid = threadIdx.x;
  const int wave = tid >> 5, ln = tid & 31, l16 = ln & 15, hi = ln >> 4;
  const int m0 = blockIdx.y * 128;
  const int n0 = blockIdx.x * 64;
  const int mi = wave & 3, ns = wave >> 2;

  v8f acc[4] = {};

  for (int kk = 0; kk < DM; kk += 32) {
    __syncthreads();
#pragma unroll
    for (int i = 0; i < 2; ++i) {   // A tile: 128x32 bf16 (512 x 16B)
      const int slot = tid + i * 256;
      const int row = slot >> 2, c = (slot & 3) << 3;
      const __bf16* src = An + (size_t)(m0 + row) * DM + kk + c;
      __builtin_prefetch(src + 32, 0, 1);
      *(uint4*)(At + row * 32 + c) = *(const uint4*)src;
    }
#pragma unroll
    for (int i = 0; i < 2; ++i) {   // W tile: 64x32 f32 -> bf16
      const int slot = tid + i * 256;
      const int row = slot >> 3, c4 = (slot & 7) << 2;
      float4 vw = *(const float4*)(ow + (size_t)(n0 + row) * DM + kk + c4);
      v4bf bw;
      bw[0] = (__bf16)vw.x; bw[1] = (__bf16)vw.y;
      bw[2] = (__bf16)vw.z; bw[3] = (__bf16)vw.w;
      *(v4bf*)(Wt + row * 32 + c4) = bw;
    }
    __syncthreads();

    const v16bf a0 = frag_a(At, 32, mi * 32 + l16, 0, hi);
    const v16bf a1 = frag_a(At, 32, mi * 32 + 16 + l16, 0, hi);
    const v16bf b0 = frag_b(Wt, 32, ns * 32 + l16, 0, hi);
    const v16bf b1 = frag_b(Wt, 32, ns * 32 + 16 + l16, 0, hi);
    acc[0] = wmma_bf16(a0, b0, acc[0]);
    acc[1] = wmma_bf16(a0, b1, acc[1]);
    acc[2] = wmma_bf16(a1, b0, acc[2]);
    acc[3] = wmma_bf16(a1, b1, acc[3]);
  }

#pragma unroll
  for (int bi = 0; bi < 2; ++bi) {
    const int o = n0 + ns * 32 + bi * 16 + l16;
#pragma unroll
    for (int ai = 0; ai < 2; ++ai)
#pragma unroll
      for (int r = 0; r < 8; ++r) {
        const int M = m0 + mi * 32 + ai * 16 + r + (hi << 3);
        out[(size_t)M * DM + o] = acc[ai * 2 + bi][r];
      }
  }
}

// ---------------------------------------------------------------------------
extern "C" void kernel_launch(void* const* d_in, const int* in_sizes, int n_in,
                              void* d_out, int out_size, void* d_ws, size_t ws_size,
                              hipStream_t stream) {
  (void)in_sizes; (void)n_in; (void)out_size; (void)ws_size;
  const float* x     = (const float*)d_in[0];
  const int*   tok   = (const int*)d_in[1];
  const float* qkv_w = (const float*)d_in[2];
  const float* o_w   = (const float*)d_in[3];
  float* out = (float*)d_out;

  // Workspace: 4 x 4Mi bf16 buffers = 32 MB total.
  unsigned short* Qb = (unsigned short*)d_ws;
  unsigned short* Kb = Qb + (size_t)4 * 1024 * 1024;
  unsigned short* Vt = Kb + (size_t)4 * 1024 * 1024;
  unsigned short* An = Vt + (size_t)4 * 1024 * 1024;

  qkv_rope_kernel<<<dim3(48, 32), 256, 0, stream>>>(x, tok, qkv_w, Qb, Kb, Vt);
  attn_kernel<<<dim3(32, 32), 128, 0, stream>>>(Qb, Kb, Vt, An);
  oproj_kernel<<<dim3(16, 32), 256, 0, stream>>>(An, o_w, out);
}
